// ModifiedAttention_56401510531281
// MI455X (gfx1250) — matrix operand order repaired and checked by reading, not compiled, not run on hardware
//
#include <hip/hip_runtime.h>
#include <hip/hip_bf16.h>

typedef __attribute__((ext_vector_type(16))) _Float16 v16h;
typedef __attribute__((ext_vector_type(8)))  _Float16 v8h;
typedef __attribute__((ext_vector_type(8)))  float    v8f;

#define NB    64
#define NN    257
#define CC    768
#define NH    12
#define HD2   64
#define NP    288           // padded key length: 18 tiles of 16
#define QPR   272           // padded query rows: 17 tiles of 16
#define MROWS (NB * NN)     // 16448 (divisible by 16)
#define QT    17            // query tiles (272 rows cover 257)

// ---------- fragment helpers (layouts per CDNA5 ISA 7.12.2, wave32) ----------

// A fragment 16x32 f16: lane m = lane&15, half = lane>>4.
// In-lane elements: K = k0 + half*8 + 0..7  and  K = k0 + 16 + half*8 + 0..7.
static __device__ inline v16h load_afrag(const _Float16* base, int ld, int row,
                                         int k0, int half) {
  const _Float16* p = base + (size_t)row * ld + k0 + half * 8;
  v8h x0 = *(const v8h*)p;
  v8h x1 = *(const v8h*)(p + 16);
  return __builtin_shufflevector(x0, x1, 0, 1, 2, 3, 4, 5, 6, 7,
                                 8, 9, 10, 11, 12, 13, 14, 15);
}

// B fragment 32x16 f16 where B[k][n] = src[n*ld + k0 + k]:
// lane n = lane&15 holds K = k0 + half*8 + 0..7 and K = k0 + 16 + half*8 + 0..7.
static __device__ inline v16h load_bfrag(const _Float16* base, int ld, int n,
                                         int k0, int half) {
  const _Float16* p = base + (size_t)n * ld + k0 + half * 8;
  v8h x0 = *(const v8h*)p;
  v8h x1 = *(const v8h*)(p + 16);
  return __builtin_shufflevector(x0, x1, 0, 1, 2, 3, 4, 5, 6, 7,
                                 8, 9, 10, 11, 12, 13, 14, 15);
}

static __device__ inline v8f wmma_f16(v16h a, v16h b, v8f c) {
  return __builtin_amdgcn_wmma_f32_16x16x32_f16(false, a, false, b,
                                                (short)0, c, false, false);
}

static __device__ inline v8f vzero8() {
  v8f z = {0.f, 0.f, 0.f, 0.f, 0.f, 0.f, 0.f, 0.f};
  return z;
}

// ---------- kernel 1: fp32 -> f16 conversion of x, qkv_w, proj_w ----------

__global__ void prep_f16(const float* __restrict__ x,
                         const float* __restrict__ w1,
                         const float* __restrict__ w2,
                         _Float16* __restrict__ xh,
                         _Float16* __restrict__ wh1,
                         _Float16* __restrict__ wh2) {
  size_t i = (size_t)blockIdx.x * blockDim.x + threadIdx.x;
  size_t stride = (size_t)gridDim.x * blockDim.x;
  const size_t nx = (size_t)MROWS * CC;
  const size_t n1 = (size_t)3 * CC * CC;
  const size_t n2 = (size_t)CC * CC;
  for (size_t t = i; t < nx; t += stride) xh[t] = (_Float16)x[t];
  for (size_t t = i; t < n1; t += stride) wh1[t] = (_Float16)w1[t];
  for (size_t t = i; t < n2; t += stride) wh2[t] = (_Float16)w2[t];
}

// ---------- kernel 1b: materialize rel-pos bias[h][qrow][key] once ----------
// Bias depends only on (h, qrow, key): gathering it here (instead of inside
// the attention kernel) removes the dependent rel->rpb gather chain and the
// 64x per-batch redundancy. 12*272*288 f16 = 1.9 MB, L2-resident.

__global__ void bias_prep(const float* __restrict__ rpb,
                          const int* __restrict__ rel,
                          _Float16* __restrict__ biasb) {
  const size_t total = (size_t)NH * QPR * NP;
  size_t i = (size_t)blockIdx.x * blockDim.x + threadIdx.x;
  size_t stride = (size_t)gridDim.x * blockDim.x;
  for (size_t t = i; t < total; t += stride) {
    int k = (int)(t % NP);
    int r = (int)((t / NP) % QPR);
    int h = (int)(t / ((size_t)NP * QPR));
    int qr = r < NN ? r : NN - 1;
    float v = (k < NN) ? rpb[rel[qr * NN + k] * NH + h] : 0.f;
    biasb[t] = (_Float16)v;
  }
}

// ---------- kernel 2: QKV GEMM + bias + scale + scatter to q/k/vT ----------
// out[gm][gc] = sum_k xh[gm][k] * qkv_w[gc][k]  (gc in [0,2304))
// Per wave: 16 rows x 64 cols (4 accumulators). K-loop is ping-pong
// double-buffered: two named fragment sets, each reloaded directly (no
// register rotation copies) while the other set's 4 WMMAs issue.

__global__ void __launch_bounds__(128) qkv_gemm(
    const _Float16* __restrict__ xh, const _Float16* __restrict__ wh1,
    const float* __restrict__ q_bias, const float* __restrict__ v_bias,
    _Float16* __restrict__ qb, _Float16* __restrict__ kb,
    _Float16* __restrict__ vtb) {
  const int lane = threadIdx.x & 31, wave = threadIdx.x >> 5;
  const int half = lane >> 4, nl = lane & 15;
  const int mtile = blockIdx.x;
  const int cg = blockIdx.y * 4 + wave;  // 0..35 column-group of 64
  const int arow = mtile * 16 + nl;

  v8f acc[4];
  for (int i = 0; i < 4; ++i) acc[i] = vzero8();

  // prologue: preload k-chunks 0 and 32 into the two sets
  v16h a0 = load_afrag(xh, CC, arow, 0, half);
  v16h b0[4], b1[4];
#pragma unroll
  for (int dt = 0; dt < 4; ++dt)
    b0[dt] = load_bfrag(wh1, CC, (cg * 4 + dt) * 16 + nl, 0, half);
  v16h a1 = load_afrag(xh, CC, arow, 32, half);
#pragma unroll
  for (int dt = 0; dt < 4; ++dt)
    b1[dt] = load_bfrag(wh1, CC, (cg * 4 + dt) * 16 + nl, 32, half);

#pragma unroll 1
  for (int k0 = 0; k0 + 64 < CC; k0 += 64) {
#pragma unroll
    for (int dt = 0; dt < 4; ++dt) acc[dt] = wmma_f16(a0, b0[dt], acc[dt]);
    a0 = load_afrag(xh, CC, arow, k0 + 64, half);
#pragma unroll
    for (int dt = 0; dt < 4; ++dt)
      b0[dt] = load_bfrag(wh1, CC, (cg * 4 + dt) * 16 + nl, k0 + 64, half);
#pragma unroll
    for (int dt = 0; dt < 4; ++dt) acc[dt] = wmma_f16(a1, b1[dt], acc[dt]);
    a1 = load_afrag(xh, CC, arow, k0 + 96, half);
#pragma unroll
    for (int dt = 0; dt < 4; ++dt)
      b1[dt] = load_bfrag(wh1, CC, (cg * 4 + dt) * 16 + nl, k0 + 96, half);
  }
  // tail: consume final two chunks (704, 736)
#pragma unroll
  for (int dt = 0; dt < 4; ++dt) acc[dt] = wmma_f16(a0, b0[dt], acc[dt]);
#pragma unroll
  for (int dt = 0; dt < 4; ++dt) acc[dt] = wmma_f16(a1, b1[dt], acc[dt]);

#pragma unroll
  for (int j = 0; j < 8; ++j) {
    const int gm = mtile * 16 + j + 8 * half;  // < 16448 always
    const int bb = gm / NN, nn = gm % NN;
#pragma unroll
    for (int dt = 0; dt < 4; ++dt) {
      const int gc = (cg * 4 + dt) * 16 + nl;
      const int kind = gc / CC, rem = gc % CC;
      const int h = rem >> 6, d = rem & 63;
      float val = acc[dt][j];
      if (kind == 0) {
        val = (val + q_bias[rem]) * 0.125f;  // SCALE = 64^-0.5
        qb[(((size_t)bb * NH + h) * NP + nn) * HD2 + d] = (_Float16)val;
      } else if (kind == 1) {
        kb[(((size_t)bb * NH + h) * NP + nn) * HD2 + d] = (_Float16)val;
      } else {
        val += v_bias[rem];
        vtb[(((size_t)bb * NH + h) * HD2 + d) * NP + nn] = (_Float16)val;
      }
    }
  }
}

// ---------- kernel 3: flash attention per (b,h), one wave per 16-query tile ----

__global__ void __launch_bounds__(128) attn_kernel(
    const _Float16* __restrict__ qb, const _Float16* __restrict__ kb,
    const _Float16* __restrict__ vtb, const _Float16* __restrict__ biasb,
    _Float16* __restrict__ ao) {
  __shared__ __align__(32) _Float16 plds[4][16 * 32];
  const int lane = threadIdx.x & 31, wave = threadIdx.x >> 5;
  const int half = lane >> 4, nl = lane & 15;
  const int bh = blockIdx.x;
  const int b = bh / NH, h = bh % NH;
  const int qt = blockIdx.y * 4 + wave;
  if (qt >= QT) return;

  const _Float16* qp = qb + (size_t)bh * NP * HD2;
  const _Float16* kp = kb + (size_t)bh * NP * HD2;
  const _Float16* vp = vtb + (size_t)bh * HD2 * NP;
  const _Float16* bp = biasb + (size_t)h * QPR * NP;  // bias[h][qrow][key]
  _Float16* pl = plds[wave];

  // Q fragments (16 rows x 64 dims) pinned in registers for the whole pass.
  const v16h qa0 = load_afrag(qp, HD2, qt * 16 + nl, 0, half);
  const v16h qa1 = load_afrag(qp, HD2, qt * 16 + nl, 32, half);

  v8f acc[4];
  for (int i = 0; i < 4; ++i) acc[i] = vzero8();
  float mrun[8], lsum[8];
#pragma unroll
  for (int j = 0; j < 8; ++j) { mrun[j] = -3.0e38f; lsum[j] = 0.f; }

  for (int kt = 0; kt < NP / 32; ++kt) {
    const int key0 = kt * 32;
    // logits: two 16x16 tiles covering keys [key0, key0+32)
    v8f s0 = vzero8(), s1 = vzero8();
    v16h kb00 = load_bfrag(kp, HD2, key0 + nl, 0, half);
    v16h kb01 = load_bfrag(kp, HD2, key0 + nl, 32, half);
    v16h kb10 = load_bfrag(kp, HD2, key0 + 16 + nl, 0, half);
    v16h kb11 = load_bfrag(kp, HD2, key0 + 16 + nl, 32, half);
    s0 = wmma_f16(qa0, kb00, s0);
    s0 = wmma_f16(qa1, kb01, s0);
    s1 = wmma_f16(qa0, kb10, s1);
    s1 = wmma_f16(qa1, kb11, s1);

    const int ka = key0 + nl, kc = key0 + 16 + nl;
#pragma unroll
    for (int j = 0; j < 8; ++j) {
      const int qrow = qt * 16 + j + 8 * half;  // <= 271 < QPR
      const _Float16* br = bp + (size_t)qrow * NP;
      // coalesced f16 bias loads (key index == lane index within tile)
      float v0 = (ka < NN) ? s0[j] + (float)br[ka] : -1.0e30f;
      float v1 = (kc < NN) ? s1[j] + (float)br[kc] : -1.0e30f;
      // row max across the 16 lanes of this half (C layout: row spans lanes)
      float t = fmaxf(v0, v1);
      t = fmaxf(t, __shfl_xor(t, 1, 32));
      t = fmaxf(t, __shfl_xor(t, 2, 32));
      t = fmaxf(t, __shfl_xor(t, 4, 32));
      t = fmaxf(t, __shfl_xor(t, 8, 32));
      const float mnew = fmaxf(mrun[j], t);
      const float alpha = __expf(mrun[j] - mnew);
      mrun[j] = mnew;
      acc[0][j] *= alpha; acc[1][j] *= alpha;
      acc[2][j] *= alpha; acc[3][j] *= alpha;
      const float p0 = __expf(v0 - mnew);
      const float p1 = __expf(v1 - mnew);
      float rs = p0 + p1;
      rs += __shfl_xor(rs, 1, 32);
      rs += __shfl_xor(rs, 2, 32);
      rs += __shfl_xor(rs, 4, 32);
      rs += __shfl_xor(rs, 8, 32);
      lsum[j] = lsum[j] * alpha + rs;
      const int prow = j + 8 * half;
      pl[prow * 32 + nl] = (_Float16)p0;        // C layout -> LDS
      pl[prow * 32 + 16 + nl] = (_Float16)p1;
    }
    asm volatile("s_wait_dscnt 0" ::: "memory");  // own-wave LDS store->load fence

    // P (16x32) as A fragment from LDS, then P @ V over 4 d-tiles.
    v16h pa = load_afrag(pl, 32, nl, 0, half);
#pragma unroll
    for (int dt = 0; dt < 4; ++dt) {
      v16h vb = load_bfrag(vp, NP, dt * 16 + nl, key0, half);
      acc[dt] = wmma_f16(pa, vb, acc[dt]);
    }
  }

#pragma unroll
  for (int j = 0; j < 8; ++j) {
    const int qrow = qt * 16 + j + 8 * half;
    if (qrow >= NN) continue;
    const float inv = 1.0f / lsum[j];
#pragma unroll
    for (int dt = 0; dt < 4; ++dt)
      ao[((size_t)b * NN + qrow) * CC + h * HD2 + dt * 16 + nl] =
          (_Float16)(acc[dt][j] * inv);
  }
}

// ---------- kernel 4: output projection GEMM + bias (fp32 out) ----------
// Same ping-pong double-buffered structure as qkv_gemm.

__global__ void __launch_bounds__(128) proj_gemm(
    const _Float16* __restrict__ ao, const _Float16* __restrict__ wh2,
    const float* __restrict__ proj_b, float* __restrict__ out) {
  const int lane = threadIdx.x & 31, wave = threadIdx.x >> 5;
  const int half = lane >> 4, nl = lane & 15;
  const int mtile = blockIdx.x;
  const int cg = blockIdx.y * 4 + wave;  // 0..11
  const int arow = mtile * 16 + nl;

  v8f acc[4];
  for (int i = 0; i < 4; ++i) acc[i] = vzero8();

  v16h a0 = load_afrag(ao, CC, arow, 0, half);
  v16h b0[4], b1[4];
#pragma unroll
  for (int dt = 0; dt < 4; ++dt)
    b0[dt] = load_bfrag(wh2, CC, (cg * 4 + dt) * 16 + nl, 0, half);
  v16h a1 = load_afrag(ao, CC, arow, 32, half);
#pragma unroll
  for (int dt = 0; dt < 4; ++dt)
    b1[dt] = load_bfrag(wh2, CC, (cg * 4 + dt) * 16 + nl, 32, half);

#pragma unroll 1
  for (int k0 = 0; k0 + 64 < CC; k0 += 64) {
#pragma unroll
    for (int dt = 0; dt < 4; ++dt) acc[dt] = wmma_f16(a0, b0[dt], acc[dt]);
    a0 = load_afrag(ao, CC, arow, k0 + 64, half);
#pragma unroll
    for (int dt = 0; dt < 4; ++dt)
      b0[dt] = load_bfrag(wh2, CC, (cg * 4 + dt) * 16 + nl, k0 + 64, half);
#pragma unroll
    for (int dt = 0; dt < 4; ++dt) acc[dt] = wmma_f16(a1, b1[dt], acc[dt]);
    a1 = load_afrag(ao, CC, arow, k0 + 96, half);
#pragma unroll
    for (int dt = 0; dt < 4; ++dt)
      b1[dt] = load_bfrag(wh2, CC, (cg * 4 + dt) * 16 + nl, k0 + 96, half);
  }
#pragma unroll
  for (int dt = 0; dt < 4; ++dt) acc[dt] = wmma_f16(a0, b0[dt], acc[dt]);
#pragma unroll
  for (int dt = 0; dt < 4; ++dt) acc[dt] = wmma_f16(a1, b1[dt], acc[dt]);

#pragma unroll
  for (int j = 0; j < 8; ++j) {
    const int gm = mtile * 16 + j + 8 * half;
#pragma unroll
    for (int dt = 0; dt < 4; ++dt) {
      const int gc = (cg * 4 + dt) * 16 + nl;
      out[(size_t)gm * CC + gc] = acc[dt][j] + proj_b[gc];
    }
  }
}

// ---------- launch ----------

extern "C" void kernel_launch(void* const* d_in, const int* in_sizes, int n_in,
                              void* d_out, int out_size, void* d_ws,
                              size_t ws_size, hipStream_t stream) {
  const float* x      = (const float*)d_in[0];
  const float* qkv_w  = (const float*)d_in[1];
  const float* q_bias = (const float*)d_in[2];
  const float* v_bias = (const float*)d_in[3];
  const float* rpb    = (const float*)d_in[4];
  const float* proj_w = (const float*)d_in[5];
  const float* proj_b = (const float*)d_in[6];
  const int*   rel    = (const int*)d_in[7];
  float* out = (float*)d_out;

  _Float16* ws = (_Float16*)d_ws;
  size_t o = 0;
  _Float16* xh  = ws + o; o += (size_t)MROWS * CC;      // x in f16 (reused as ao)
  _Float16* wh1 = ws + o; o += (size_t)3 * CC * CC;     // qkv_w f16
  _Float16* wh2 = ws + o; o += (size_t)CC * CC;         // proj_w f16
  _Float16* qb  = ws + o; o += (size_t)NB * NH * NP * HD2;  // Q  [bh][n][d]
  _Float16* kb  = ws + o; o += (size_t)NB * NH * NP * HD2;  // K  [bh][n][d]
  _Float16* vtb = ws + o; o += (size_t)NB * NH * HD2 * NP;  // V^T [bh][d][n]
  _Float16* biasb = ws + o; o += (size_t)NH * QPR * NP;     // bias[h][q][k]
  _Float16* ao  = xh;  // alias: x f16 dead after qkv_gemm

  // zero q/k/vT pads (keys/rows >= 257) so padded lanes stay finite
  hipMemsetAsync(qb, 0, (size_t)3 * NB * NH * NP * HD2 * sizeof(_Float16),
                 stream);

  prep_f16<<<2048, 256, 0, stream>>>(x, qkv_w, proj_w, xh, wh1, wh2);
  bias_prep<<<2048, 256, 0, stream>>>(rpb, rel, biasb);
  qkv_gemm<<<dim3(MROWS / 16, 9), 128, 0, stream>>>(xh, wh1, q_bias, v_bias,
                                                    qb, kb, vtb);
  attn_kernel<<<dim3(NB * NH, 5), 128, 0, stream>>>(qb, kb, vtb, biasb, ao);
  proj_gemm<<<dim3(MROWS / 16, 3), 128, 0, stream>>>(ao, wh2, proj_b, out);
}
